// CausalSelfAttentionMLA_4380866642222
// MI455X (gfx1250) — matrix operand order repaired and checked by reading, not compiled, not run on hardware
//
#include <hip/hip_runtime.h>
#include <hip/hip_bf16.h>

// MLA causal attention for gfx1250: all GEMMs via v_wmma_f32_16x16x32_bf16.
// B=2, T=2048, C=2048, H=16, Dh=128, Dl=64.
// Fragments are filled by pairs of b128 loads; pure-copy LDS staging uses
// global_load_async_to_lds_b128 (ASYNCcnt).

#define B_  2
#define T_  2048
#define C_  2048
#define H_  16
#define DH_ 128
#define DL_ 64

typedef __attribute__((ext_vector_type(16))) __bf16        v16bf;
typedef __attribute__((ext_vector_type(8)))  float         v8f;
typedef __attribute__((ext_vector_type(8)))  unsigned int  v8u;
typedef __attribute__((ext_vector_type(4)))  unsigned int  v4u;

union Frag {
  v16bf h;
  v8u   u;
  v4u   q[2];
  unsigned int w[8];
};

__device__ __forceinline__ unsigned int pack_bf16(float a, float b) {
  union { __bf16 h[2]; unsigned int u; } p;
  p.h[0] = (__bf16)a;   // low half  = even k
  p.h[1] = (__bf16)b;   // high half = odd k
  return p.u;
}

__device__ __forceinline__ v8f wmma_bf16(v16bf a, v16bf b, v8f c) {
  return __builtin_amdgcn_wmma_f32_16x16x32_bf16(false, a, false, b, (short)0, c,
                                                 false, false);
}

// Async global->LDS 16-byte copy (CDNA5, ASYNCcnt). Per-lane addresses.
__device__ __forceinline__ void async_copy_b128(void* lds_dst, const void* g_src) {
  const unsigned lds = (unsigned)(uintptr_t)lds_dst;  // low 32b of generic ptr = LDS addr
  const unsigned long long ga = (unsigned long long)(uintptr_t)g_src;
  asm volatile("global_load_async_to_lds_b128 %0, %1, off" :: "v"(lds), "v"(ga) : "memory");
}
__device__ __forceinline__ void wait_async() {
  asm volatile("s_wait_asynccnt 0x0" ::: "memory");
}

// A fragment (16x32 bf16): element (m,k) = src[m*ld + k].
// Lanes 0-15 -> M=lane, K=0..15; lanes 16-31 -> K=16..31.
// Two b128 loads; requires ld multiple of 8 bf16 and 16B-aligned base.
__device__ __forceinline__ v16bf load_a_frag(const __bf16* src, int ld) {
  const int lane = threadIdx.x & 31;
  const int hi   = lane >> 4;
  const int m    = lane & 15;
  const unsigned int* p = (const unsigned int*)src + m * (ld >> 1) + (hi << 3);
  Frag f;
  f.q[0] = *(const v4u*)(p);
  f.q[1] = *(const v4u*)(p + 4);
  return f.h;
}

// B fragment (32x16 bf16) from a fragment-major uint tile:
// srcU[n*16 + kp] holds k=2*kp (low) / 2*kp+1 (high). Lane n reads kp=hi*8+0..7
// -> 8 consecutive uints -> two b128 loads.
__device__ __forceinline__ v16bf load_b_frag_fm(const unsigned int* srcU) {
  const int lane = threadIdx.x & 31;
  const int hi   = lane >> 4;
  const int n    = lane & 15;
  const v4u* p = (const v4u*)(srcU + n * 16 + (hi << 3));
  Frag f;
  f.q[0] = p[0];
  f.q[1] = p[1];
  return f.h;
}

// B fragment from a transposed source: element (k,n) = src[n*ld + k].
// Lane n reads k=hi*16+0..15 -> 8 consecutive uints -> two b128 loads.
__device__ __forceinline__ v16bf load_b_frag_T(const __bf16* src, int ld) {
  const int lane = threadIdx.x & 31;
  const int hi   = lane >> 4;
  const int n    = lane & 15;
  const v4u* p = (const v4u*)((const unsigned int*)src + n * (ld >> 1) + (hi << 3));
  Frag f;
  f.q[0] = p[0];
  f.q[1] = p[1];
  return f.h;
}

// RoPE on a C-fragment element: column n sits in lane (n&15); pair partner = lane^1.
__device__ __forceinline__ float rope_apply(float v, int d, int t, int dim) {
  const int i = d >> 1;
  const float ang = (float)t * __expf(-9.210340371976184f * (float)(2 * i) / (float)dim);
  float s, c;
  __sincosf(ang, &s, &c);
  const float other = __shfl_xor(v, 1, 32);
  return v * c + ((d & 1) ? other * s : -other * s);
}

// ---------------------------------------------------------------- q = x@wq + RoPE
__global__ __launch_bounds__(256) void k_qproj_rope(const float* __restrict__ x,
                                                    const float* __restrict__ wq,
                                                    __bf16* __restrict__ qbf) {
  __shared__ alignas(16) __bf16       As[128][40];
  __shared__ alignas(16) unsigned int BsU[8][16][16];
  const int tid = threadIdx.x, wid = tid >> 5, lane = tid & 31;
  const int hi = lane >> 4, col = lane & 15;
  const int mBase = blockIdx.x * 128;
  const int nBase = blockIdx.y * 128;
  const int rA = tid >> 4, cA = (tid & 15) * 2;   // A-staging home (8 rows, stride 16)
  const int kpB = tid >> 7, cB = tid & 127;       // B-staging home (8 kp, stride 2)

  v8f acc[8] = {};
  for (int kb = 0; kb < C_; kb += 32) {
    const float* xs = &x[(size_t)(mBase + rA) * C_ + kb + cA];
#pragma unroll
    for (int it = 0; it < 8; ++it)
      *(unsigned int*)&As[rA + it * 16][cA] =
          pack_bf16(xs[(size_t)it * 16 * C_], xs[(size_t)it * 16 * C_ + 1]);
    const float* ws = &wq[(size_t)(kb + 2 * kpB) * (H_ * DH_) + nBase + cB];
#pragma unroll
    for (int it = 0; it < 8; ++it)
      BsU[cB >> 4][cB & 15][kpB + it * 2] =
          pack_bf16(ws[(size_t)it * 4 * (H_ * DH_)], ws[(size_t)it * 4 * (H_ * DH_) + (H_ * DH_)]);
    if (kb + 32 < C_) {  // global_prefetch_b8 of next k-slice
      __builtin_prefetch(&x[(size_t)(mBase + (tid >> 1)) * C_ + kb + 32 + (tid & 1) * 16], 0, 1);
      __builtin_prefetch(&wq[(size_t)(kb + 32 + (tid >> 3)) * (H_ * DH_) + nBase + (tid & 7) * 16], 0, 1);
    }
    __syncthreads();
    const v16bf a = load_a_frag(&As[wid * 16][0], 40);
#pragma unroll
    for (int ns = 0; ns < 8; ++ns) {
      const v16bf b = load_b_frag_fm(&BsU[ns][0][0]);
      acc[ns] = wmma_bf16(a, b, acc[ns]);
    }
    __syncthreads();
  }
#pragma unroll
  for (int ns = 0; ns < 8; ++ns) {
#pragma unroll
    for (int r = 0; r < 8; ++r) {
      const int row = (hi << 3) + r;
      const int mg  = mBase + wid * 16 + row;
      const int t   = mg & (T_ - 1);
      const int bb  = mg >> 11;
      const int cg  = nBase + ns * 16 + col;
      const int h   = cg >> 7;
      const int d   = cg & (DH_ - 1);
      const float o = rope_apply(acc[ns][r], d, t, DH_);
      qbf[((bb * H_ + h) * T_ + t) * DH_ + d] = (__bf16)o;
    }
  }
}

// -------------------------------------------- k_lat = x@wk (+RoPE), v_lat = x@wv
__global__ __launch_bounds__(256) void k_latent_rope(const float* __restrict__ x,
                                                     const float* __restrict__ wk,
                                                     const float* __restrict__ wv,
                                                     __bf16* __restrict__ klat,
                                                     __bf16* __restrict__ vlat) {
  __shared__ alignas(16) __bf16       As[128][40];
  __shared__ alignas(16) unsigned int BsU[8][16][16];
  const int tid = threadIdx.x, wid = tid >> 5, lane = tid & 31;
  const int hi = lane >> 4, col = lane & 15;
  const int mBase = blockIdx.x * 128;
  const int rA = tid >> 4, cA = (tid & 15) * 2;
  const int kpB = tid >> 7, cB = tid & 127;

  v8f acc[8] = {};
  for (int kb = 0; kb < C_; kb += 32) {
    const float* xs = &x[(size_t)(mBase + rA) * C_ + kb + cA];
#pragma unroll
    for (int it = 0; it < 8; ++it)
      *(unsigned int*)&As[rA + it * 16][cA] =
          pack_bf16(xs[(size_t)it * 16 * C_], xs[(size_t)it * 16 * C_ + 1]);
    const float* wsrc = (cB < DL_) ? &wk[(size_t)(kb + 2 * kpB) * DL_ + cB]
                                   : &wv[(size_t)(kb + 2 * kpB) * DL_ + cB - DL_];
#pragma unroll
    for (int it = 0; it < 8; ++it)
      BsU[cB >> 4][cB & 15][kpB + it * 2] =
          pack_bf16(wsrc[(size_t)it * 4 * DL_], wsrc[(size_t)it * 4 * DL_ + DL_]);
    if (kb + 32 < C_)
      __builtin_prefetch(&x[(size_t)(mBase + (tid >> 1)) * C_ + kb + 32 + (tid & 1) * 16], 0, 1);
    __syncthreads();
    const v16bf a = load_a_frag(&As[wid * 16][0], 40);
#pragma unroll
    for (int ns = 0; ns < 8; ++ns) {
      const v16bf b = load_b_frag_fm(&BsU[ns][0][0]);
      acc[ns] = wmma_bf16(a, b, acc[ns]);
    }
    __syncthreads();
  }
#pragma unroll
  for (int ns = 0; ns < 8; ++ns) {
#pragma unroll
    for (int r = 0; r < 8; ++r) {
      const int row = (hi << 3) + r;
      const int mg  = mBase + wid * 16 + row;
      const int t   = mg & (T_ - 1);
      const int cg  = ns * 16 + col;
      const float val = acc[ns][r];
      if (cg < DL_) {  // uniform per ns
        const float o = rope_apply(val, cg, t, DL_);
        klat[mg * DL_ + cg] = (__bf16)o;
      } else {
        vlat[mg * DL_ + (cg - DL_)] = (__bf16)val;
      }
    }
  }
}

// ------------------------------------- k/v = lat @ expand[h]  (K=64, per head)
// K path -> [b][h][t][d] row-major. V path -> fragment-major
// uint[(bh)][t/32][d/16][n=d%16][kp=(t%32)/2] so P*V B-frags are two b128 loads.
__global__ __launch_bounds__(256) void k_expand(const __bf16* __restrict__ klat,
                                                const __bf16* __restrict__ vlat,
                                                const float* __restrict__ kew,
                                                const float* __restrict__ vew,
                                                __bf16* __restrict__ kbf,
                                                __bf16* __restrict__ vfm) {
  __shared__ alignas(16) __bf16       As[128][40];
  __shared__ alignas(16) unsigned int BsU[8][16][16];
  const int tid = threadIdx.x, wid = tid >> 5, lane = tid & 31;
  const int hi = lane >> 4, col = lane & 15;
  const int mBase = blockIdx.x * 128;
  const int h = blockIdx.y;
  const bool isV = (blockIdx.z != 0);
  const __bf16* lat = isV ? vlat : klat;
  const float*  ew  = isV ? vew : kew;
  const int kpB = tid >> 7, cB = tid & 127;

  v8f acc[8] = {};
  for (int kb = 0; kb < DL_; kb += 32) {
    // A tile is a pure bf16 copy -> async global->LDS b128 (ASYNCcnt)
#pragma unroll
    for (int cch = 0; cch < 2; ++cch) {
      const int chunk = tid + (cch << 8);            // 512 chunks of 16B
      const int r = chunk >> 2, q4 = chunk & 3;
      async_copy_b128(&As[r][q4 * 8], &lat[(size_t)(mBase + r) * DL_ + kb + q4 * 8]);
    }
    const float* wsB = &ew[(size_t)(h * DL_ + kb + 2 * kpB) * DH_ + cB];
#pragma unroll
    for (int it = 0; it < 8; ++it)
      BsU[cB >> 4][cB & 15][kpB + it * 2] =
          pack_bf16(wsB[(size_t)it * 4 * DH_], wsB[(size_t)it * 4 * DH_ + DH_]);
    wait_async();
    __syncthreads();
    const v16bf a = load_a_frag(&As[wid * 16][0], 40);
#pragma unroll
    for (int ns = 0; ns < 8; ++ns) {
      const v16bf b = load_b_frag_fm(&BsU[ns][0][0]);
      acc[ns] = wmma_bf16(a, b, acc[ns]);
    }
    __syncthreads();
  }
#pragma unroll
  for (int ns = 0; ns < 8; ++ns) {
#pragma unroll
    for (int r = 0; r < 8; ++r) {
      const int row = (hi << 3) + r;
      const int mg  = mBase + wid * 16 + row;
      const int t   = mg & (T_ - 1);
      const int bb  = mg >> 11;
      const int d   = ns * 16 + col;
      const size_t head = (size_t)(bb * H_ + h) * T_ * DH_;
      if (isV) {
        const unsigned int uidx =
            ((unsigned)((t >> 5) * 8 + (d >> 4)) << 8) + ((d & 15) << 4) + ((t & 31) >> 1);
        vfm[head + ((size_t)uidx << 1) + (t & 1)] = (__bf16)acc[ns][r];
      } else {
        kbf[head + (size_t)t * DH_ + d] = (__bf16)acc[ns][r];
      }
    }
  }
}

// ----------------------------------------------------- flash attention (causal)
__global__ __launch_bounds__(256) void k_flash(const __bf16* __restrict__ q,
                                               const __bf16* __restrict__ k,
                                               const __bf16* __restrict__ vfm,
                                               __bf16* __restrict__ y) {
  __shared__ alignas(16) __bf16 Ps[8][16][72];  // per-wave P (C-frag -> A-frag)
  const int tid = threadIdx.x, wid = tid >> 5, lane = tid & 31;
  const int hi = lane >> 4, col = lane & 15;
  const int h  = blockIdx.y;
  const int bb = blockIdx.z;
  const int qBase = blockIdx.x * 128 + wid * 16;

  const size_t headOff = (size_t)(bb * H_ + h) * T_ * DH_;
  const __bf16* qh = q + headOff;
  const __bf16* kh = k + headOff;
  const unsigned int* vhU = (const unsigned int*)vfm + (headOff >> 1);

  v16bf qa[4];
#pragma unroll
  for (int kc = 0; kc < 4; ++kc)
    qa[kc] = load_a_frag(qh + (size_t)qBase * DH_ + kc * 32, DH_);

  v8f o[8] = {};
  float mRow[8], lRow[8];
#pragma unroll
  for (int r = 0; r < 8; ++r) { mRow[r] = -3.0e38f; lRow[r] = 0.f; }

  const float scale = 0.08838834764831845f;  // 1/sqrt(128)
  const int lastRow = qBase + 15;

  for (int jb = 0; jb <= lastRow; jb += 64) {
    if (jb + 64 <= lastRow)
      __builtin_prefetch(kh + (size_t)(jb + 64 + (lane >> 1)) * DH_ + wid * 16, 0, 1);
    v8f s[4] = {};
#pragma unroll
    for (int kc = 0; kc < 4; ++kc) {
#pragma unroll
      for (int ns = 0; ns < 4; ++ns) {
        const v16bf b = load_b_frag_T(kh + (size_t)(jb + ns * 16) * DH_ + kc * 32, DH_);
        s[ns] = wmma_bf16(qa[kc], b, s[ns]);
      }
    }
#pragma unroll
    for (int ns = 0; ns < 4; ++ns)
#pragma unroll
      for (int r = 0; r < 8; ++r) {
        const int tk = jb + ns * 16 + col;
        const int tq = qBase + (hi << 3) + r;
        const float val = s[ns][r] * scale;
        s[ns][r] = (tk > tq) ? -1.0e30f : val;
      }
    float alpha[8];
#pragma unroll
    for (int r = 0; r < 8; ++r) {
      float mx = fmaxf(fmaxf(s[0][r], s[1][r]), fmaxf(s[2][r], s[3][r]));
#pragma unroll
      for (int off = 1; off <= 8; off <<= 1)
        mx = fmaxf(mx, __shfl_xor(mx, off, 32));
      const float mNew = fmaxf(mRow[r], mx);
      alpha[r] = __expf(mRow[r] - mNew);
      mRow[r] = mNew;
    }
#pragma unroll
    for (int ns = 0; ns < 4; ++ns)
#pragma unroll
      for (int r = 0; r < 8; ++r)
        s[ns][r] = __expf(s[ns][r] - mRow[r]);
#pragma unroll
    for (int r = 0; r < 8; ++r) {
      float sum = s[0][r] + s[1][r] + s[2][r] + s[3][r];
#pragma unroll
      for (int off = 1; off <= 8; off <<= 1)
        sum += __shfl_xor(sum, off, 32);
      lRow[r] = lRow[r] * alpha[r] + sum;
    }
#pragma unroll
    for (int no = 0; no < 8; ++no)
#pragma unroll
      for (int r = 0; r < 8; ++r)
        o[no][r] *= alpha[r];
#pragma unroll
    for (int ns = 0; ns < 4; ++ns)
#pragma unroll
      for (int r = 0; r < 8; ++r)
        Ps[wid][(hi << 3) + r][ns * 16 + col] = (__bf16)s[ns][r];
    asm volatile("s_wait_dscnt 0x0" ::: "memory");
#pragma unroll
    for (int kc = 0; kc < 2; ++kc) {
      const v16bf pa = load_a_frag(&Ps[wid][0][kc * 32], 72);
      const int t32 = (jb + kc * 32) >> 5;
#pragma unroll
      for (int no = 0; no < 8; ++no) {
        const v16bf vb = load_b_frag_fm(vhU + ((size_t)(t32 * 8 + no) << 8));
        o[no] = wmma_bf16(pa, vb, o[no]);
      }
    }
  }
  __bf16* yb = y + (size_t)bb * T_ * C_ + h * DH_;
#pragma unroll
  for (int no = 0; no < 8; ++no)
#pragma unroll
    for (int r = 0; r < 8; ++r) {
      const int t = qBase + (hi << 3) + r;
      yb[(size_t)t * C_ + no * 16 + col] = (__bf16)(o[no][r] / lRow[r]);
    }
}

// ----------------------------------------------------------- out = y @ proj_w
__global__ __launch_bounds__(256) void k_proj(const __bf16* __restrict__ ybf,
                                              const float* __restrict__ pw,
                                              float* __restrict__ out) {
  __shared__ alignas(16) __bf16       As[128][40];
  __shared__ alignas(16) unsigned int BsU[8][16][16];
  const int tid = threadIdx.x, wid = tid >> 5, lane = tid & 31;
  const int hi = lane >> 4, col = lane & 15;
  const int mBase = blockIdx.x * 128;
  const int nBase = blockIdx.y * 128;
  const int kpB = tid >> 7, cB = tid & 127;

  v8f acc[8] = {};
  for (int kb = 0; kb < C_; kb += 32) {
    // A tile is a pure bf16 copy -> async global->LDS b128 (ASYNCcnt)
#pragma unroll
    for (int cch = 0; cch < 2; ++cch) {
      const int chunk = tid + (cch << 8);
      const int r = chunk >> 2, q4 = chunk & 3;
      async_copy_b128(&As[r][q4 * 8], &ybf[(size_t)(mBase + r) * C_ + kb + q4 * 8]);
    }
    const float* wsB = &pw[(size_t)(kb + 2 * kpB) * C_ + nBase + cB];
#pragma unroll
    for (int it = 0; it < 8; ++it)
      BsU[cB >> 4][cB & 15][kpB + it * 2] =
          pack_bf16(wsB[(size_t)it * 4 * C_], wsB[(size_t)it * 4 * C_ + C_]);
    if (kb + 32 < C_) {
      __builtin_prefetch(&ybf[(size_t)(mBase + (tid >> 1)) * C_ + kb + 32 + (tid & 1) * 16], 0, 1);
      __builtin_prefetch(&pw[(size_t)(kb + 32 + (tid >> 3)) * C_ + nBase + (tid & 7) * 16], 0, 1);
    }
    wait_async();
    __syncthreads();
    const v16bf a = load_a_frag(&As[wid * 16][0], 40);
#pragma unroll
    for (int ns = 0; ns < 8; ++ns) {
      const v16bf b = load_b_frag_fm(&BsU[ns][0][0]);
      acc[ns] = wmma_bf16(a, b, acc[ns]);
    }
    __syncthreads();
  }
#pragma unroll
  for (int ns = 0; ns < 8; ++ns)
#pragma unroll
    for (int r = 0; r < 8; ++r) {
      const int row = (hi << 3) + r;
      out[(size_t)(mBase + wid * 16 + row) * C_ + nBase + ns * 16 + col] = acc[ns][r];
    }
}

extern "C" void kernel_launch(void* const* d_in, const int* in_sizes, int n_in,
                              void* d_out, int out_size, void* d_ws, size_t ws_size,
                              hipStream_t stream) {
  const float* x   = (const float*)d_in[0];
  const float* wq  = (const float*)d_in[1];
  const float* wk  = (const float*)d_in[2];
  const float* wv  = (const float*)d_in[3];
  const float* kex = (const float*)d_in[4];
  const float* vex = (const float*)d_in[5];
  const float* pw  = (const float*)d_in[6];
  float* out = (float*)d_out;

  char* ws = (char*)d_ws;
  size_t off = 0;
  auto carve = [&](size_t bytes) {
    char* p = ws + off;
    off += (bytes + 255) & ~(size_t)255;
    return p;
  };
  __bf16* qbf  = (__bf16*)carve((size_t)B_ * H_ * T_ * DH_ * sizeof(__bf16));
  __bf16* kbf  = (__bf16*)carve((size_t)B_ * H_ * T_ * DH_ * sizeof(__bf16));
  __bf16* vfm  = (__bf16*)carve((size_t)B_ * H_ * T_ * DH_ * sizeof(__bf16));
  __bf16* klat = (__bf16*)carve((size_t)B_ * T_ * DL_ * sizeof(__bf16));
  __bf16* vlat = (__bf16*)carve((size_t)B_ * T_ * DL_ * sizeof(__bf16));
  __bf16* ybf  = (__bf16*)carve((size_t)B_ * T_ * C_ * sizeof(__bf16));
  (void)ws_size; (void)in_sizes; (void)n_in; (void)out_size;

  k_qproj_rope<<<dim3((B_ * T_) / 128, (H_ * DH_) / 128, 1), 256, 0, stream>>>(x, wq, qbf);
  k_latent_rope<<<dim3((B_ * T_) / 128, 1, 1), 256, 0, stream>>>(x, wk, wv, klat, vlat);
  k_expand<<<dim3((B_ * T_) / 128, H_, 2), 256, 0, stream>>>(klat, vlat, kex, vex, kbf, vfm);
  k_flash<<<dim3(T_ / 128, H_, B_), 256, 0, stream>>>(qbf, kbf, vfm, ybf);
  k_proj<<<dim3((B_ * T_) / 128, C_ / 128, 1), 256, 0, stream>>>(ybf, pw, out);
}
